// SelfAttention_90701119357021
// MI455X (gfx1250) — matrix operand order repaired and checked by reading, not compiled, not run on hardware
//
#include <hip/hip_runtime.h>
#include <cstdint>

#define B_    4
#define C_    256
#define N_    4096
#define KVT   64
#define QROWS 128
#define NWAVES 8
#define PSTR  72   // padded P-staging row stride (halves)

typedef _Float16 h8   __attribute__((ext_vector_type(8)));
typedef _Float16 h16  __attribute__((ext_vector_type(16)));
typedef float    f8   __attribute__((ext_vector_type(8)));
typedef int      v4i  __attribute__((ext_vector_type(4)));
typedef unsigned int u32x4 __attribute__((ext_vector_type(4)));
typedef int      i32x4 __attribute__((ext_vector_type(4)));
typedef int      i32x8 __attribute__((ext_vector_type(8)));

#define WMMA16(a, b, c) \
  __builtin_amdgcn_wmma_f32_16x16x32_f16(false, (a), false, (b), (short)0, (c), false, false)

// ---------------- CDNA5 feature detection ----------------
#if defined(__has_builtin)
#if __has_builtin(__builtin_amdgcn_tensor_load_to_lds) && \
    __has_builtin(__builtin_amdgcn_s_wait_tensorcnt)
#define USE_TDM 1
#endif
#if __has_builtin(__builtin_amdgcn_global_load_async_to_lds_b128)
#define USE_ASYNC_LDS 1
#endif
#endif

#ifdef USE_ASYNC_LDS
typedef __attribute__((address_space(1))) v4i* gptr4;
typedef __attribute__((address_space(3))) v4i* lptr4;
#endif

__device__ __forceinline__ void g2l_16B(const _Float16* g, _Float16* l) {
#ifdef USE_ASYNC_LDS
  __builtin_amdgcn_global_load_async_to_lds_b128((gptr4)(uintptr_t)g, (lptr4)l, 0, 0);
#else
  *(h8*)l = *(const h8*)g;
#endif
}

#ifdef USE_TDM
__device__ __forceinline__ unsigned lds_off(_Float16* p) {
  return (unsigned)(uintptr_t)(__attribute__((address_space(3))) _Float16*)p;
}

// 3-D tensor DMA: LDS written sequentially (z,y,x); element = 2 bytes.
// global addr of element = base + 2*(x + y*stride0 + z*stride1)
__device__ __forceinline__ void tdm_load_3d(const void* gp, unsigned ldsoff,
                                            int tile0, int tile1, int tile2,
                                            unsigned stride0, unsigned stride1) {
  unsigned long long ga = (unsigned long long)(uintptr_t)gp;
  u32x4 g0;
  g0[0] = 1u;                                                 // count=1, user mode
  g0[1] = ldsoff;                                             // lds_addr (bytes)
  g0[2] = (unsigned)ga;                                       // global_addr[31:0]
  g0[3] = ((unsigned)(ga >> 32) & 0x01FFFFFFu) | (2u << 30);  // ga[56:32] | type=2
  i32x8 g1;
  g1[0] = (int)(1u << 16);                                    // data_size=1 (2B)
  g1[1] = 0;                                                  // tensor_dim0[15:0]=0 (dim0=2^30)
  g1[2] = (int)0x00004000;                                    // tensor_dim0[31:16] ; tensor_dim1[15:0]=0
  g1[3] = (int)(0x00004000u | ((unsigned)tile0 << 16));       // tensor_dim1[31:16] ; tile_dim0
  g1[4] = (int)(((unsigned)tile2 << 16) | (unsigned)tile1);   // tile_dim1 ; tile_dim2
  g1[5] = (int)stride0;                                       // tensor_dim0_stride[31:0]
  g1[6] = (int)((stride1 & 0xFFFFu) << 16);                   // s0[47:32]=0 ; s1[15:0]
  g1[7] = (int)(stride1 >> 16);                               // s1[47:16]
  i32x4 g2 = {0x40000000, 0, 0, 0};                           // tensor_dim2 big; rest 0
  i32x4 g3 = {0, 0, 0, 0};
  i32x8 g4 = {0, 0, 0, 0, 0, 0, 0, 0};                        // unused extra group (6-arg form)
  __builtin_amdgcn_tensor_load_to_lds(g0, g1, g2, g3, g4, 0);
}
#endif

// Stage one KV tile into chunked LDS layouts:
//   Kl[(mt*256 + c)*16 + j] = K[c][m1 + mt*16 + j]   (kg = &K[0][m1], row stride N)
//   Vl[(ct*64 + m)*16 + j]  = V[m1 + m][ct*16 + j]   (vg = &V[m1][0], row stride C)
__device__ __forceinline__ void stage_kv(const _Float16* kg, const _Float16* vg,
                                         _Float16* Kl, _Float16* Vl, int tid, int wave) {
#ifdef USE_TDM
  if (wave == 0) {
    tdm_load_3d(kg, lds_off(Kl), 16, 256, 4, 4096u, 16u);
    tdm_load_3d(vg, lds_off(Vl), 16, 64, 16, 256u, 16u);
  }
  (void)tid;
#else
#pragma unroll
  for (int mt = 0; mt < 4; ++mt) {  // K: tid = c row
    const _Float16* s = kg + (size_t)tid * N_ + mt * 16;
    _Float16* d = Kl + (mt * 256 + tid) * 16;
    g2l_16B(s, d);
    g2l_16B(s + 8, d + 8);
  }
  const int m = tid & 63, gq = tid >> 6;
#pragma unroll
  for (int j = 0; j < 4; ++j) {     // V: 4 ct chunks per thread
    const int ct = gq * 4 + j;
    const _Float16* s = vg + m * C_ + ct * 16;
    _Float16* d = Vl + (ct * 64 + m) * 16;
    g2l_16B(s, d);
    g2l_16B(s + 8, d + 8);
  }
#endif
}

__device__ __forceinline__ void stage_wait(int wave) {
#if defined(USE_TDM)
  if (wave == 0) __builtin_amdgcn_s_wait_tensorcnt(0);
#elif defined(USE_ASYNC_LDS)
#if __has_builtin(__builtin_amdgcn_s_wait_asynccnt)
  __builtin_amdgcn_s_wait_asynccnt(0);
#else
  asm volatile("s_wait_asynccnt 0" ::: "memory");
#endif
#endif
}

// A-operand gather: halves 0..7 at p, halves 8..15 at p+16 (ISA 16-bit A layout)
__device__ __forceinline__ h16 ld16a(const _Float16* p) {
  h8 lo = *(const h8*)p;
  h8 hi = *(const h8*)(p + 16);
  h16 r;
#pragma unroll
  for (int i = 0; i < 8; ++i) { r[i] = lo[i]; r[i + 8] = hi[i]; }
  return r;
}

// B-operand gather: lane = N column; K rows 0..7 at p, 16..23 at p+16*st (stride st)
__device__ __forceinline__ h16 ld16c(const _Float16* p, int st) {
  h16 r;
#pragma unroll
  for (int i = 0; i < 8; ++i) { r[i] = p[i * st]; r[i + 8] = p[(16 + i) * st]; }
  return r;
}

// ================= Kernel 0: f32 -> f16 bulk convert =================
__global__ void __launch_bounds__(256) cvt_f16(const float* __restrict__ src,
                                               _Float16* __restrict__ dst, int n8) {
  const int i = blockIdx.x * 256 + threadIdx.x;
  if (i < n8) {
    const float4 a = *(const float4*)(src + (size_t)i * 8);
    const float4 b = *(const float4*)(src + (size_t)i * 8 + 4);
    h8 o;
    o[0] = (_Float16)a.x; o[1] = (_Float16)a.y; o[2] = (_Float16)a.z; o[3] = (_Float16)a.w;
    o[4] = (_Float16)b.x; o[5] = (_Float16)b.y; o[6] = (_Float16)b.z; o[7] = (_Float16)b.w;
    *(h8*)(dst + (size_t)i * 8) = o;
  }
}

// ================= Kernel 1: Q/K/V projections (1x1 conv) =================
// D[o][n] = sum_c W[o][c] * x[b][c][n] + bias[o]; all operands already f16.
__global__ void __launch_bounds__(256) proj_qkv(
    const _Float16* __restrict__ x16,   // (B, C, N) f16
    const _Float16* __restrict__ Wq16, const float* __restrict__ bq,
    const _Float16* __restrict__ Wk16, const float* __restrict__ bk,
    const _Float16* __restrict__ Wv16, const float* __restrict__ bv,
    _Float16* __restrict__ q16,   // (B, N, C)
    _Float16* __restrict__ k16,   // (B, C, N)
    _Float16* __restrict__ v16)   // (B, N, C)
{
  const int lane = threadIdx.x & 31;
  const int lrow = lane & 15, lhi = lane >> 4;
  const int job  = blockIdx.x * NWAVES + (threadIdx.x >> 5);
  const int proj = job / (B_ * (N_ / 16));
  const int rem  = job % (B_ * (N_ / 16));
  const int b    = rem >> 8;
  const int n0   = (rem & 255) << 4;

  const _Float16* W;
  const float* bias;
  if (proj == 0)      { W = Wq16; bias = bq; }
  else if (proj == 1) { W = Wk16; bias = bk; }
  else                { W = Wv16; bias = bv; }

  h16 bx[8];  // B-operand chunks of x: 32(c) x 16(n), lane = n column
#pragma unroll
  for (int kc = 0; kc < 8; ++kc)
    bx[kc] = ld16c(x16 + ((size_t)b * C_ + kc * 32 + lhi * 8) * N_ + n0 + lrow, N_);

  for (int ot = 0; ot < 16; ++ot) {
    const int o0 = ot << 4;
    f8 acc = {0.f, 0.f, 0.f, 0.f, 0.f, 0.f, 0.f, 0.f};
#pragma unroll
    for (int kc = 0; kc < 8; ++kc) {
      h16 a = ld16a(W + (size_t)(o0 + lrow) * C_ + kc * 32 + lhi * 8);
      acc = WMMA16(a, bx[kc], acc);
    }
#pragma unroll
    for (int v = 0; v < 8; ++v) acc[v] += bias[o0 + 8 * lhi + v];

    if (proj == 1) {  // K: (b, c, m) — scattered 2B stores (row stride N)
#pragma unroll
      for (int v = 0; v < 8; ++v)
        k16[((size_t)b * C_ + o0 + 8 * lhi + v) * N_ + n0 + lrow] = (_Float16)acc[v];
    } else {          // Q / V: (b, n, c) — contiguous 16B store
      _Float16* dst = (proj == 0 ? q16 : v16) +
                      ((size_t)b * N_ + n0 + lrow) * C_ + o0 + 8 * lhi;
      h8 st;
#pragma unroll
      for (int v = 0; v < 8; ++v) st[v] = (_Float16)acc[v];
      *(h8*)dst = st;
    }
  }
}

// ================= Kernel 2: flash attention =================
__global__ void __launch_bounds__(256) attn_flash(
    const _Float16* __restrict__ q16,  // (B, N, C)
    const _Float16* __restrict__ k16,  // (B, C, N)
    const _Float16* __restrict__ v16,  // (B, N, C)
    float* __restrict__ out)           // (B, C, N)
{
  __shared__ _Float16 Kb[2][4 * 256 * 16];   // chunk layout: (mt, c, 16)
  __shared__ _Float16 Vb[2][16 * 64 * 16];   // chunk layout: (ct, m, 16)
  __shared__ _Float16 Pb[NWAVES][16 * PSTR];

  const int tid  = threadIdx.x;
  const int lane = tid & 31, wave = tid >> 5;
  const int lrow = lane & 15, lhi = lane >> 4;
  const int b    = blockIdx.x >> 5;
  const int nW   = (blockIdx.x & 31) * QROWS + wave * 16;

  const _Float16* kbase = k16 + (size_t)b * C_ * N_;
  const _Float16* vbase = v16 + (size_t)b * N_ * C_;

  h16 qa[8];  // Q tile (16 x 256) resident as A-operand chunks
#pragma unroll
  for (int kc = 0; kc < 8; ++kc)
    qa[kc] = ld16a(q16 + ((size_t)b * N_ + nW + lrow) * C_ + kc * 32 + lhi * 8);

  f8 O[16];
#pragma unroll
  for (int ct = 0; ct < 16; ++ct)
#pragma unroll
    for (int v = 0; v < 8; ++v) O[ct][v] = 0.f;
  float mi[8], li[8];
#pragma unroll
  for (int v = 0; v < 8; ++v) { mi[v] = -1e30f; li[v] = 0.f; }

  stage_kv(kbase, vbase, Kb[0], Vb[0], tid, wave);  // prologue: tile 0

#pragma unroll 2
  for (int it = 0; it < N_ / KVT; ++it) {
    const int cur = it & 1;
    stage_wait(wave);
    __syncthreads();

    if (it + 1 < N_ / KVT)
      stage_kv(kbase + (it + 1) * KVT, vbase + (size_t)(it + 1) * KVT * C_,
               Kb[(it + 1) & 1], Vb[(it + 1) & 1], tid, wave);

    // ---- S = Q . K^T  (16 x 64), 32 WMMAs ----
    f8 S[4];
#pragma unroll
    for (int mt = 0; mt < 4; ++mt) {
      f8 s = {0.f, 0.f, 0.f, 0.f, 0.f, 0.f, 0.f, 0.f};
#pragma unroll
      for (int kc = 0; kc < 8; ++kc) {
        h16 kb = ld16c(&Kb[cur][(mt * 256 + kc * 32 + lhi * 8) * 16 + lrow], 16);
        s = WMMA16(qa[kc], kb, s);
      }
      S[mt] = s;
    }

    // ---- online softmax (rows live in 16-lane halves) ----
    float alpha[8];
#pragma unroll
    for (int v = 0; v < 8; ++v) {
      float mx = fmaxf(fmaxf(S[0][v], S[1][v]), fmaxf(S[2][v], S[3][v]));
#pragma unroll
      for (int off = 8; off >= 1; off >>= 1)
        mx = fmaxf(mx, __shfl_xor(mx, off, 32));
      float mn = fmaxf(mi[v], mx);
      alpha[v] = __expf(mi[v] - mn);
      mi[v] = mn;
    }
    float rs[8];
#pragma unroll
    for (int v = 0; v < 8; ++v) rs[v] = 0.f;
#pragma unroll
    for (int mt = 0; mt < 4; ++mt)
#pragma unroll
      for (int v = 0; v < 8; ++v) {
        float p = __expf(S[mt][v] - mi[v]);
        S[mt][v] = p;
        rs[v] += p;
      }
#pragma unroll
    for (int v = 0; v < 8; ++v) {
      float r = rs[v];
#pragma unroll
      for (int off = 8; off >= 1; off >>= 1) r += __shfl_xor(r, off, 32);
      li[v] = li[v] * alpha[v] + r;
    }
#pragma unroll
    for (int ct = 0; ct < 16; ++ct)
#pragma unroll
      for (int v = 0; v < 8; ++v) O[ct][v] *= alpha[v];

    // ---- P: D-layout -> A-layout via padded per-wave LDS tile ----
#pragma unroll
    for (int mt = 0; mt < 4; ++mt)
#pragma unroll
      for (int v = 0; v < 8; ++v)
        Pb[wave][(v + 8 * lhi) * PSTR + mt * 16 + lrow] = (_Float16)S[mt][v];

    // ---- O += P . V  (16 x 256), 32 WMMAs ----
#pragma unroll
    for (int kc2 = 0; kc2 < 2; ++kc2) {
      h16 pa = ld16a(&Pb[wave][lrow * PSTR + kc2 * 32 + lhi * 8]);
#pragma unroll
      for (int ct = 0; ct < 16; ++ct) {
        h16 vb = ld16c(&Vb[cur][(ct * 64 + kc2 * 32 + lhi * 8) * 16 + lrow], 16);
        O[ct] = WMMA16(pa, vb, O[ct]);
      }
    }
  }

  // ---- epilogue: normalize and store out (b, c, n) ----
  float inv[8];
#pragma unroll
  for (int v = 0; v < 8; ++v) inv[v] = 1.0f / li[v];
#pragma unroll
  for (int ct = 0; ct < 16; ++ct) {
    float* op = out + ((size_t)b * C_ + ct * 16 + lrow) * N_ + nW + 8 * lhi;
    float4 s0 = {O[ct][0] * inv[0], O[ct][1] * inv[1], O[ct][2] * inv[2], O[ct][3] * inv[3]};
    float4 s1 = {O[ct][4] * inv[4], O[ct][5] * inv[5], O[ct][6] * inv[6], O[ct][7] * inv[7]};
    *(float4*)op = s0;
    *(float4*)(op + 4) = s1;
  }
}

// ================= host launcher =================
extern "C" void kernel_launch(void* const* d_in, const int* in_sizes, int n_in,
                              void* d_out, int out_size, void* d_ws, size_t ws_size,
                              hipStream_t stream) {
  (void)in_sizes; (void)n_in; (void)out_size; (void)ws_size;
  const float* x  = (const float*)d_in[0];
  const float* Wq = (const float*)d_in[1];
  const float* bq = (const float*)d_in[2];
  const float* Wk = (const float*)d_in[3];
  const float* bk = (const float*)d_in[4];
  const float* Wv = (const float*)d_in[5];
  const float* bv = (const float*)d_in[6];
  float* out = (float*)d_out;

  const size_t TENS = (size_t)B_ * N_ * C_;  // 4,194,304 halves
  _Float16* q16  = (_Float16*)d_ws;
  _Float16* k16  = q16 + TENS;
  _Float16* v16  = k16 + TENS;
  _Float16* x16  = v16 + TENS;
  _Float16* Wq16 = x16 + TENS;
  _Float16* Wk16 = Wq16 + C_ * C_;
  _Float16* Wv16 = Wk16 + C_ * C_;

  cvt_f16<<<dim3((int)(TENS / 8 / 256)), dim3(256), 0, stream>>>(x, x16, (int)(TENS / 8));
  cvt_f16<<<dim3(C_ * C_ / 8 / 256), dim3(256), 0, stream>>>(Wq, Wq16, C_ * C_ / 8);
  cvt_f16<<<dim3(C_ * C_ / 8 / 256), dim3(256), 0, stream>>>(Wk, Wk16, C_ * C_ / 8);
  cvt_f16<<<dim3(C_ * C_ / 8 / 256), dim3(256), 0, stream>>>(Wv, Wv16, C_ * C_ / 8);
  proj_qkv<<<dim3(3 * B_ * (N_ / 16) / NWAVES), dim3(256), 0, stream>>>(
      x16, Wq16, bq, Wk16, bk, Wv16, bv, q16, k16, v16);
  attn_flash<<<dim3(B_ * (N_ / QROWS)), dim3(256), 0, stream>>>(q16, k16, v16, out);
}
